// GCN_81655918232022
// MI455X (gfx1250) — matrix operand order repaired and checked by reading, not compiled, not run on hardware
//
#include <hip/hip_runtime.h>
#include <math.h>

#define NNODES   92160
#define NEDGES   2949120
#define NGRAPH   256
#define NODESPG  360
#define FIN      360
#define HDIM     64
#define HIDDIM   128

typedef __attribute__((ext_vector_type(16))) __bf16       v16bf;
typedef __attribute__((ext_vector_type(8)))  float        v8f;
typedef __attribute__((ext_vector_type(8)))  unsigned int v8u;
typedef __attribute__((ext_vector_type(4)))  unsigned int v4u;

__device__ __forceinline__ unsigned short f2bf(float f) {
  unsigned u = __builtin_bit_cast(unsigned, f);
  return (unsigned short)((u + 0x7FFFu + ((u >> 16) & 1u)) >> 16);  // RNE
}

// ---------------- pack weights into WMMA B-fragment layout (contiguous k order, same as A staging) ----------------
// P[((kt*ntiles + nt)*32 + lane)*8 + v] = bf16 pair W[k][n], W[k+1][n],
//   k = kt*32 + 16*half + 2v, n = nt*16 + (lane&15); zero beyond K.
__global__ void k_pack_w(const float* __restrict__ W, unsigned int* __restrict__ P,
                         int K, int N, int total) {
  int o = blockIdx.x * blockDim.x + threadIdx.x;
  if (o >= total) return;
  const int v    = o & 7;
  const int lane = (o >> 3) & 31;
  const int tile = o >> 8;
  const int ntiles = N >> 4;
  const int kt = tile / ntiles;
  const int nt = tile - kt * ntiles;
  const int l16 = lane & 15, half = lane >> 4;
  const int k = kt * 32 + half * 16 + v * 2;
  const int n = nt * 16 + l16;
  unsigned lo = (k     < K) ? (unsigned)f2bf(W[(size_t)k       * N + n]) : 0u;
  unsigned hi = (k + 1 < K) ? (unsigned)f2bf(W[(size_t)(k + 1) * N + n]) : 0u;
  P[o] = lo | (hi << 16);
}

// ---------------- degree / dinv ----------------
__global__ void k_fill(float* __restrict__ p, float v, int n) {
  int i = blockIdx.x * blockDim.x + threadIdx.x;
  if (i < n) p[i] = v;
}
__global__ void k_deg_count(const int* __restrict__ dst, float* __restrict__ deg, int e) {
  int i = blockIdx.x * blockDim.x + threadIdx.x;
  if (i < e)
    __hip_atomic_fetch_add(&deg[dst[i]], 1.0f, __ATOMIC_RELAXED, __HIP_MEMORY_SCOPE_AGENT);
}
__global__ void k_rsqrt_inplace(float* __restrict__ p, int n) {
  int i = blockIdx.x * blockDim.x + threadIdx.x;
  if (i < n) { float v = p[i]; p[i] = (v > 0.0f) ? rsqrtf(v) : 0.0f; }
}

// ---------------- WMMA GEMM: C[M x NCOL] = A[M x KREAL](f32) * Wpacked, optional BN+ReLU ----
// KT = KREAL rounded up to 32. Fully unrolled K loop; only the tail tile (layer 0) masks.
template <int KT, int KREAL, int NCOL, bool BN>
__global__ __launch_bounds__(256) void k_gemm(
    const float* __restrict__ A, const unsigned int* __restrict__ Bp,
    float* __restrict__ C, int M,
    const float* __restrict__ bias, const float* __restrict__ gamma,
    const float* __restrict__ beta, const float* __restrict__ mean,
    const float* __restrict__ var)
{
  (void)M;
  constexpr int NWAVES = NCOL / 16;
  constexpr int MWAVES = 8 / NWAVES;
  constexpr int BM     = MWAVES * 16;
  constexpr int KTILES = KT / 32;
  __shared__ alignas(16) unsigned short lds[BM][40];  // 80B stride: 16B-aligned b128 reads

  const int tid  = threadIdx.x;
  const int wave = tid >> 5, lane = tid & 31;
  const int half = lane >> 4, l16 = lane & 15;
  const int mwave = wave / NWAVES;
  const int nwave = wave - mwave * NWAVES;
  const int blockRow = blockIdx.x * BM;

  v8f acc = {0.f, 0.f, 0.f, 0.f, 0.f, 0.f, 0.f, 0.f};
  const unsigned int* bptr = Bp + ((size_t)nwave * 32 + lane) * 8;

  #pragma unroll
  for (int kt = 0; kt < KTILES; ++kt) {
    const int k0 = kt * 32;
    // cooperative stage: BM x 32 fp32 -> bf16 pairs in LDS (coalesced 64B/row, no branches)
    auto stage = [&](int u) {
      const int r  = u >> 4;
      const int c2 = u & 15;
      const int kk = k0 + (c2 << 1);
      const float* ap = A + (size_t)(blockRow + r) * KREAL + kk;
      float lo, hi;
      if ((KT == KREAL) || (k0 + 31 < KREAL)) {        // compile-time per iteration
        const float2 v2 = *(const float2*)ap;
        lo = v2.x; hi = v2.y;
        if (kt + 1 < KTILES) __builtin_prefetch(ap + 32, 0, 3);
      } else {                                          // tail tile only (layer 0)
        lo = (kk     < KREAL) ? ap[0] : 0.f;
        hi = (kk + 1 < KREAL) ? ap[1] : 0.f;
      }
      *(unsigned int*)&lds[r][c2 << 1] =
          (unsigned)f2bf(lo) | ((unsigned)f2bf(hi) << 16);
    };
    stage(tid);
    if constexpr (BM == 32) stage(tid + 256);
    __syncthreads();

    // A fragment: two contiguous 16B chunks per lane (au[0..3] at k=16h, au[4..7] at k=16h+8)
    const unsigned short* ap = &lds[(mwave << 4) + l16][half << 4];
    const v4u alo = *(const v4u*)ap;
    const v4u ahi = *(const v4u*)(ap + 8);
    const v8u au  = __builtin_shufflevector(alo, ahi, 0, 1, 2, 3, 4, 5, 6, 7);
    // B fragment: one coalesced 32B per-lane load of pre-packed weights
    const v8u bu  = *(const v8u*)(bptr + (size_t)kt * (NWAVES * 256));

    acc = __builtin_amdgcn_wmma_f32_16x16x32_bf16(
        false, __builtin_bit_cast(v16bf, au),
        false, __builtin_bit_cast(v16bf, bu),
        (short)0, acc, false, false);
    __syncthreads();
  }

  #pragma unroll
  for (int v = 0; v < 8; ++v) {
    const int row = blockRow + (mwave << 4) + (half << 3) + v;
    const int col = (nwave << 4) + l16;
    float val = acc[v];
    if constexpr (BN) {  // fused bias + BatchNorm(eval) + ReLU
      val = (val + bias[col] - mean[col]) * rsqrtf(var[col] + 1e-5f) * gamma[col] + beta[col];
      val = fmaxf(val, 0.0f);
    }
    C[(size_t)row * NCOL + col] = val;
  }
}

// ---------------- GCN message passing ----------------
__global__ void k_self_init(const float* __restrict__ L, const float* __restrict__ dinv,
                            float* __restrict__ out, int total) {
  int i = blockIdx.x * blockDim.x + threadIdx.x;
  if (i < total) {
    const int n = i >> 6;
    const float di = dinv[n];
    out[i] = L[i] * di * di;   // self-loop contribution
  }
}
__global__ void k_edge_scatter(const float* __restrict__ L, const int* __restrict__ src,
                               const int* __restrict__ dst, const float* __restrict__ dinv,
                               float* __restrict__ out, int e) {
  int t = blockIdx.x * blockDim.x + threadIdx.x;
  int ei = t >> 5, l = t & 31;           // one wave32 covers 64 features as float2
  if (ei >= e) return;
  const int s = src[ei], d = dst[ei];
  const float w = dinv[s] * dinv[d];
  const float2 hv = *(const float2*)(L + (size_t)s * HDIM + l * 2);
  float* o = out + (size_t)d * HDIM + l * 2;
  __hip_atomic_fetch_add(o,     hv.x * w, __ATOMIC_RELAXED, __HIP_MEMORY_SCOPE_AGENT);
  __hip_atomic_fetch_add(o + 1, hv.y * w, __ATOMIC_RELAXED, __HIP_MEMORY_SCOPE_AGENT);
}
__global__ void k_bias_relu(float* __restrict__ h, const float* __restrict__ b, int total) {
  int i = blockIdx.x * blockDim.x + threadIdx.x;
  if (i < total) h[i] = fmaxf(h[i] + b[i & (HDIM - 1)], 0.0f);
}

// ---------------- global_add_pool ----------------
__global__ __launch_bounds__(256) void k_pool(const float* __restrict__ h, float* __restrict__ p) {
  __shared__ float red[256];
  const int g = blockIdx.x;
  const int f = threadIdx.x & 63;
  const int c = threadIdx.x >> 6;
  float s = 0.f;
  for (int i = c; i < NODESPG; i += 4)
    s += h[((size_t)g * NODESPG + i) * HDIM + f];
  red[threadIdx.x] = s;
  __syncthreads();
  if (c == 0) p[g * HDIM + f] = red[f] + red[64 + f] + red[128 + f] + red[192 + f];
}

// ---------------- final linear (64->2) + log_softmax ----------------
__global__ void k_head(const float* __restrict__ z, const float* __restrict__ w,
                       const float* __restrict__ b, float* __restrict__ out) {
  int r = blockIdx.x * blockDim.x + threadIdx.x;
  if (r >= NGRAPH) return;
  float a0 = b[0], a1 = b[1];
  #pragma unroll 8
  for (int k = 0; k < HDIM; ++k) {
    const float zv = z[(size_t)r * HDIM + k];
    a0 += zv * w[k * 2 + 0];
    a1 += zv * w[k * 2 + 1];
  }
  const float m = fmaxf(a0, a1);
  const float lse = m + logf(__expf(a0 - m) + __expf(a1 - m));
  out[r * 2 + 0] = a0 - lse;
  out[r * 2 + 1] = a1 - lse;
}

extern "C" void kernel_launch(void* const* d_in, const int* in_sizes, int n_in,
                              void* d_out, int out_size, void* d_ws, size_t ws_size,
                              hipStream_t stream) {
  (void)in_sizes; (void)n_in; (void)out_size; (void)ws_size;
  const float* x   = (const float*)d_in[0];
  const int*   ei  = (const int*)d_in[1];
  const int* srcI  = ei;
  const int* dstI  = ei + NEDGES;
  const float* w0  = (const float*)d_in[3];  const float* b0 = (const float*)d_in[4];
  const float* w1  = (const float*)d_in[5];  const float* b1 = (const float*)d_in[6];
  const float* w2  = (const float*)d_in[7];  const float* b2 = (const float*)d_in[8];
  const float* m1w = (const float*)d_in[9];  const float* m1b = (const float*)d_in[10];
  const float* g1  = (const float*)d_in[11]; const float* be1 = (const float*)d_in[12];
  const float* mu1 = (const float*)d_in[13]; const float* v1  = (const float*)d_in[14];
  const float* m2w = (const float*)d_in[15]; const float* m2b = (const float*)d_in[16];
  const float* g2  = (const float*)d_in[17]; const float* be2 = (const float*)d_in[18];
  const float* mu2 = (const float*)d_in[19]; const float* v2  = (const float*)d_in[20];
  const float* m3w = (const float*)d_in[21]; const float* m3b = (const float*)d_in[22];
  const float* g3  = (const float*)d_in[23]; const float* be3 = (const float*)d_in[24];
  const float* mu3 = (const float*)d_in[25]; const float* v3  = (const float*)d_in[26];
  const float* m4w = (const float*)d_in[27]; const float* m4b = (const float*)d_in[28];

  char* ws = (char*)d_ws;
  size_t off = 0;
  auto alloc = [&](size_t bytes) -> void* {
    void* p = ws + off;
    off = (off + bytes + 255) & ~(size_t)255;
    return p;
  };
  // packed weight fragment buffers: (Kpad/32)*(N/16)*256 uints each
  unsigned int* pw0 = (unsigned int*)alloc((size_t)12 * 4 * 256 * 4);  // K 360->384, N 64
  unsigned int* pw1 = (unsigned int*)alloc((size_t)2  * 4 * 256 * 4);  // 64x64
  unsigned int* pw2 = (unsigned int*)alloc((size_t)2  * 4 * 256 * 4);
  unsigned int* pm1 = (unsigned int*)alloc((size_t)2  * 8 * 256 * 4);  // 64x128
  unsigned int* pm2 = (unsigned int*)alloc((size_t)4  * 4 * 256 * 4);  // 128x64
  unsigned int* pm3 = (unsigned int*)alloc((size_t)2  * 4 * 256 * 4);
  float* dinv = (float*)alloc((size_t)NNODES * 4);
  float* L    = (float*)alloc((size_t)NNODES * HDIM * 4);
  float* Hb   = (float*)alloc((size_t)NNODES * HDIM * 4);
  float* pool = (float*)alloc((size_t)NGRAPH * HDIM * 4);
  float* z1   = (float*)alloc((size_t)NGRAPH * HIDDIM * 4);
  float* z2   = (float*)alloc((size_t)NGRAPH * HDIM * 4);
  float* z3   = (float*)alloc((size_t)NGRAPH * HDIM * 4);

  auto pack = [&](const float* W, unsigned int* P, int K, int N, int ktiles) {
    const int total = ktiles * (N >> 4) * 256;
    k_pack_w<<<(total + 255) / 256, 256, 0, stream>>>(W, P, K, N, total);
  };
  pack(w0,  pw0, FIN,    HDIM,   12);
  pack(w1,  pw1, HDIM,   HDIM,   2);
  pack(w2,  pw2, HDIM,   HDIM,   2);
  pack(m1w, pm1, HDIM,   HIDDIM, 2);
  pack(m2w, pm2, HIDDIM, HDIM,   4);
  pack(m3w, pm3, HDIM,   HDIM,   2);

  // degree (shared by all three conv layers): deg = 1 (self loop) + in-degree
  k_fill<<<(NNODES + 255) / 256, 256, 0, stream>>>(dinv, 1.0f, NNODES);
  k_deg_count<<<(NEDGES + 255) / 256, 256, 0, stream>>>(dstI, dinv, NEDGES);
  k_rsqrt_inplace<<<(NNODES + 255) / 256, 256, 0, stream>>>(dinv, NNODES);

  const int tot = NNODES * HDIM;
  auto msgpass = [&](const float* bias) {
    k_self_init<<<(tot + 255) / 256, 256, 0, stream>>>(L, dinv, Hb, tot);
    k_edge_scatter<<<((size_t)NEDGES * 32 + 255) / 256, 256, 0, stream>>>(
        L, srcI, dstI, dinv, Hb, NEDGES);
    k_bias_relu<<<(tot + 255) / 256, 256, 0, stream>>>(Hb, bias, tot);
  };

  // layer 0: K=360 (padded 384), layers 1/2: K=64
  k_gemm<384, 360, 64, false><<<NNODES / 32, 256, 0, stream>>>(
      x, pw0, L, NNODES, nullptr, nullptr, nullptr, nullptr, nullptr);
  msgpass(b0);
  k_gemm<64, 64, 64, false><<<NNODES / 32, 256, 0, stream>>>(
      Hb, pw1, L, NNODES, nullptr, nullptr, nullptr, nullptr, nullptr);
  msgpass(b1);
  k_gemm<64, 64, 64, false><<<NNODES / 32, 256, 0, stream>>>(
      Hb, pw2, L, NNODES, nullptr, nullptr, nullptr, nullptr, nullptr);
  msgpass(b2);

  k_pool<<<NGRAPH, 256, 0, stream>>>(Hb, pool);

  k_gemm<64, 64, 128, true><<<NGRAPH / 16, 256, 0, stream>>>(
      pool, pm1, z1, NGRAPH, m1b, g1, be1, mu1, v1);
  k_gemm<128, 128, 64, true><<<NGRAPH / 32, 256, 0, stream>>>(
      z1, pm2, z2, NGRAPH, m2b, g2, be2, mu2, v2);
  k_gemm<64, 64, 64, true><<<NGRAPH / 32, 256, 0, stream>>>(
      z2, pm3, z3, NGRAPH, m3b, g3, be3, mu3, v3);
  k_head<<<1, 256, 0, stream>>>(z3, m4w, m4b, (float*)d_out);
}
